// GTS_GCGRUCell_78400333021775
// MI455X (gfx1250) — matrix operand order repaired and checked by reading, not compiled, not run on hardware
//
#include <hip/hip_runtime.h>
#include <math.h>

// ---------------------------------------------------------------------------
// GCGRU cell (diffusion-conv GRU), MI455X / gfx1250.
// 6x GEMM  Y[2048 x 2112] = alpha*(A[2048x2048] @ X) - Xprev  via
// v_wmma_f32_16x16x32_bf16, with Tensor-Data-Mover (tensor_load_to_lds)
// double-buffered LDS staging of 128x32 A/B tiles, TENSORcnt sync, and the
// TDM pad feature (80B LDS row stride) for conflict-free ds_load_b128
// fragment reads. X matrices kept transposed (XT[c][node]) so WMMA B-frags
// and D-stores are contiguous per lane.
// ---------------------------------------------------------------------------

typedef __bf16 bf16;
typedef __attribute__((ext_vector_type(8)))  __bf16 v8bf;
typedef __attribute__((ext_vector_type(16))) __bf16 v16bf;
typedef __attribute__((ext_vector_type(8)))  float  v8f;
typedef __attribute__((ext_vector_type(4)))  unsigned int v4u;
typedef __attribute__((ext_vector_type(8)))  int v8i;
typedef __attribute__((ext_vector_type(4)))  int v4i;

#define N_     2048
#define B_     32
#define F_     66           // IN_DIM + UNITS
#define C_     2112         // B_ * F_
#define C_PAD  2176         // 17 * 128, zero-padded tail rows
#define U_     64           // UNITS
#define BNU_   4194304      // B_*N_*U_
#define CPN_   4456448      // C_PAD*N_
#define LROW   40           // LDS row stride in elems (64B data + 16B pad)

__device__ __forceinline__ v16bf join16(v8bf lo, v8bf hi) {
    v16bf r;
#pragma unroll
    for (int i = 0; i < 8; ++i) { r[i] = lo[i]; r[i + 8] = hi[i]; }
    return r;
}

// --- TDM 2D tile load: [128 rows x 32 elems] bf16, row stride 2048 elems,
// --- LDS padded to 80B/row (pad_interval=16 DWORDs, pad_amount=4 DWORDs).
__device__ __forceinline__ void tdm_load_tile(unsigned lds_addr,
                                              const bf16* gptr,
                                              unsigned tensor_d1) {
    unsigned long long ga = (unsigned long long)(uintptr_t)gptr;
    v4u g0;
    g0[0] = 1u;                                         // count=1, user mode
    g0[1] = lds_addr;                                   // LDS dest (bytes)
    g0[2] = (unsigned)(ga & 0xFFFFFFFFu);               // global_addr lo
    g0[3] = (unsigned)((ga >> 32) & 0x1FFFFFFu)         // global_addr hi
            | 0x80000000u;                              // type=2 ("image")
    v8i g1;
    g1[0] = (1 << 16)                                   // data_size = 2B
          | (1 << 20)                                   // pad_enable
          | (3 << 22)                                   // pad_interval: 16 DW
          | (3 << 25);                                  // pad_amount : 4 DW
    g1[1] = (int)((N_ & 0xFFFF) << 16);                 // tensor_dim0 lo16
    g1[2] = (int)(((unsigned)N_ >> 16) | ((tensor_d1 & 0xFFFFu) << 16));
    g1[3] = (int)((tensor_d1 >> 16) | (32u << 16));     // tile_dim0 = 32
    g1[4] = 128;                                        // tile_dim1 = 128
    g1[5] = N_;                                         // dim0_stride lo32
    g1[6] = 0;
    g1[7] = 0;
    v4i z4 = {0, 0, 0, 0};
    v8i z8 = {0, 0, 0, 0, 0, 0, 0, 0};
    __builtin_amdgcn_tensor_load_to_lds(g0, g1, z4, z4, z8, 0);
}

// --- adj f32 -> bf16 --------------------------------------------------------
__global__ __launch_bounds__(256) void cvt_adj(const float* __restrict__ s,
                                               bf16* __restrict__ a) {
    int tid = blockIdx.x * 256 + threadIdx.x;           // N_*N_ total
    a[tid] = (bf16)s[tid];
}

// --- build X0T[c][n] from inputs / h_prev (zero pad rows) ------------------
__global__ __launch_bounds__(256) void pack_x0(const float* __restrict__ in,
                                               const float* __restrict__ hp,
                                               bf16* __restrict__ x0) {
    int tid = blockIdx.x * 256 + threadIdx.x;           // CPN_ total
    int n = tid & (N_ - 1);
    int c = tid >> 11;
    float v = 0.f;
    if (c < C_) {
        int b = c / F_, f = c % F_;
        v = (f < 2) ? in[(size_t)b * N_ * 2 + n * 2 + f]
                    : hp[(size_t)b * (N_ * U_) + n * U_ + (f - 2)];
    }
    x0[tid] = (bf16)v;
}

// --- build XC0T[c][n] = concat(inputs, r*h) (zero pad rows) ----------------
__global__ __launch_bounds__(256) void pack_xc(const float* __restrict__ in,
                                               const float* __restrict__ hp,
                                               const float* __restrict__ r,
                                               bf16* __restrict__ x0) {
    int tid = blockIdx.x * 256 + threadIdx.x;           // CPN_ total
    int n = tid & (N_ - 1);
    int c = tid >> 11;
    float v = 0.f;
    if (c < C_) {
        int b = c / F_, f = c % F_;
        if (f < 2) {
            v = in[(size_t)b * N_ * 2 + n * 2 + f];
        } else {
            size_t idx = (size_t)b * (N_ * U_) + n * U_ + (f - 2);
            v = r[idx] * hp[idx];
        }
    }
    x0[tid] = (bf16)v;
}

// --- WMMA GEMM: XoutT = alpha*(A @ Xin) - Xprev  (XT layout [C_PAD][N_]) ---
// Workgroup = 8 waves = 128(M) x 128(C) tile; wave (wr,wc) in 2x4 grid does
// 64x32 via 4x2 WMMA accumulators. A/B 128x32 tiles TDM-staged into LDS,
// double buffered, issued by wave 0, synced with s_wait_tensorcnt + barrier.
__global__ __launch_bounds__(256) void gemm_cheb(const bf16* __restrict__ A,
                                                 const bf16* __restrict__ Xin,
                                                 const bf16* __restrict__ Xprev,
                                                 bf16* __restrict__ Xout,
                                                 float alpha) {
    __shared__ __align__(16) bf16 sA[2][128 * LROW];
    __shared__ __align__(16) bf16 sB[2][128 * LROW];

    const int lane   = threadIdx.x & 31;
    const int waveId = threadIdx.x >> 5;                // 0..7
    const int wr     = waveId >> 2;                     // 0..1 (M)
    const int wc     = waveId & 3;                      // 0..3 (C)
    const int mBase  = (blockIdx.x / (C_PAD / 128)) * 128;
    const int cBase  = (blockIdx.x % (C_PAD / 128)) * 128;
    const int laneLo = lane & 15;
    const int laneHi = lane >> 4;
    const int aKsel  = laneHi * 8;
    const int bKsel  = laneHi * 8;

    const v8f zero = {0.f, 0.f, 0.f, 0.f, 0.f, 0.f, 0.f, 0.f};
    v8f acc[4][2];
#pragma unroll
    for (int t = 0; t < 4; ++t)
#pragma unroll
        for (int s = 0; s < 2; ++s) acc[t][s] = zero;

    // Preload k-step 0 into buffer 0.
    if (threadIdx.x < 32) {
        tdm_load_tile((unsigned)(uintptr_t)&sA[0][0],
                      A + (size_t)mBase * N_, N_);
        tdm_load_tile((unsigned)(uintptr_t)&sB[0][0],
                      Xin + (size_t)cBase * N_, C_PAD);
        __builtin_amdgcn_s_wait_tensorcnt(0);
    }
    __syncthreads();

    int buf = 0;
    for (int kb = 0; kb < N_; kb += 32) {
        const int nxt = buf ^ 1;
        if (kb + 32 < N_ && threadIdx.x < 32) {
            tdm_load_tile((unsigned)(uintptr_t)&sA[nxt][0],
                          A + (size_t)mBase * N_ + (kb + 32), N_);
            tdm_load_tile((unsigned)(uintptr_t)&sB[nxt][0],
                          Xin + (size_t)cBase * N_ + (kb + 32), C_PAD);
        }

        v16bf afr[4];
#pragma unroll
        for (int t = 0; t < 4; ++t) {
            const bf16* p = &sA[buf][(wr * 64 + t * 16 + laneLo) * LROW + aKsel];
            afr[t] = join16(*(const v8bf*)p, *(const v8bf*)(p + 16));
        }
        v16bf bfr[2];
#pragma unroll
        for (int s = 0; s < 2; ++s) {
            const bf16* p = &sB[buf][(wc * 32 + s * 16 + laneLo) * LROW + bKsel];
            bfr[s] = join16(*(const v8bf*)p, *(const v8bf*)(p + 16));
        }
#pragma unroll
        for (int t = 0; t < 4; ++t)
#pragma unroll
            for (int s = 0; s < 2; ++s)
                acc[t][s] = __builtin_amdgcn_wmma_f32_16x16x32_bf16(
                    false, afr[t], false, bfr[s], (short)0, acc[t][s],
                    false, false);

        if (threadIdx.x < 32) __builtin_amdgcn_s_wait_tensorcnt(0);
        __syncthreads();
        buf = nxt;
    }

    // Epilogue: D lane layout -> lane writes 8 consecutive nodes of one XT row.
#pragma unroll
    for (int t = 0; t < 4; ++t)
#pragma unroll
        for (int s = 0; s < 2; ++s) {
            size_t addr = (size_t)(cBase + wc * 32 + s * 16 + laneLo) * N_ +
                          (mBase + wr * 64 + t * 16 + laneHi * 8);
            v8bf o;
            if (Xprev) {
                v8bf pv = *(const v8bf*)(Xprev + addr);
#pragma unroll
                for (int g = 0; g < 8; ++g)
                    o[g] = (bf16)(alpha * acc[t][s][g] - (float)pv[g]);
            } else {
#pragma unroll
                for (int g = 0; g < 8; ++g)
                    o[g] = (bf16)(alpha * acc[t][s][g]);
            }
            *(v8bf*)(Xout + addr) = o;
        }
}

// --- r/u gates: [65536,264] @ [264,64] (+bias, sigmoid) --------------------
__global__ __launch_bounds__(256) void gate_ru(
    const bf16* __restrict__ X0, const bf16* __restrict__ X1,
    const bf16* __restrict__ X2, const bf16* __restrict__ X3,
    const float* __restrict__ rk, const float* __restrict__ rb,
    const float* __restrict__ uk, const float* __restrict__ ub,
    float* __restrict__ rOut, float* __restrict__ uOut) {
    int tid = blockIdx.x * 256 + threadIdx.x;   // BNU_, = b*131072+n*64+u
    int u  = tid & 63;
    int bn = tid >> 6;
    int b  = bn >> 11;
    int n  = bn & (N_ - 1);
    float ra = rb[u], ua = ub[u];
    const bf16* Xs[4] = {X0, X1, X2, X3};
    for (int f = 0; f < F_; ++f) {
        size_t xa = (size_t)(b * F_ + f) * N_ + n;
#pragma unroll
        for (int m = 0; m < 4; ++m) {
            float xv = (float)Xs[m][xa];
            int row = f * 4 + m;                 // feature-major, cheb inner
            ra += xv * rk[row * U_ + u];
            ua += xv * uk[row * U_ + u];
        }
    }
    rOut[tid] = 1.f / (1.f + __expf(-ra));
    uOut[tid] = 1.f / (1.f + __expf(-ua));
}

// --- c gate + final h = u*h_prev + (1-u)*tanh(c) ---------------------------
__global__ __launch_bounds__(256) void gate_c_final(
    const bf16* __restrict__ X0, const bf16* __restrict__ X1,
    const bf16* __restrict__ X2, const bf16* __restrict__ X3,
    const float* __restrict__ ck, const float* __restrict__ cb,
    const float* __restrict__ uBuf, const float* __restrict__ hp,
    float* __restrict__ out) {
    int tid = blockIdx.x * 256 + threadIdx.x;   // BNU_
    int u  = tid & 63;
    int bn = tid >> 6;
    int b  = bn >> 11;
    int n  = bn & (N_ - 1);
    float ca = cb[u];
    const bf16* Xs[4] = {X0, X1, X2, X3};
    for (int f = 0; f < F_; ++f) {
        size_t xa = (size_t)(b * F_ + f) * N_ + n;
#pragma unroll
        for (int m = 0; m < 4; ++m) {
            float xv = (float)Xs[m][xa];
            ca += xv * ck[(f * 4 + m) * U_ + u];
        }
    }
    float c  = tanhf(ca);
    float uv = uBuf[tid];
    out[tid] = uv * hp[tid] + (1.f - uv) * c;
}

// ---------------------------------------------------------------------------
extern "C" void kernel_launch(void* const* d_in, const int* in_sizes, int n_in,
                              void* d_out, int out_size, void* d_ws,
                              size_t ws_size, hipStream_t stream) {
    const float* inputs   = (const float*)d_in[0];
    const float* supports = (const float*)d_in[1];
    const float* h_prev   = (const float*)d_in[2];
    const float* rk       = (const float*)d_in[3];
    const float* rbias    = (const float*)d_in[4];
    const float* uk       = (const float*)d_in[5];
    const float* ubias    = (const float*)d_in[6];
    const float* ck       = (const float*)d_in[7];
    const float* cbias    = (const float*)d_in[8];
    float* out = (float*)d_out;

    char* ws = (char*)d_ws;
    size_t off = 0;
    bf16* adjB = (bf16*)(ws + off); off += (size_t)N_ * N_ * sizeof(bf16);
    bf16* X[4];
    for (int i = 0; i < 4; ++i) { X[i] = (bf16*)(ws + off); off += (size_t)CPN_ * sizeof(bf16); }
    float* rBuf = (float*)(ws + off); off += (size_t)BNU_ * sizeof(float);
    float* uBuf = (float*)(ws + off); off += (size_t)BNU_ * sizeof(float);

    const int gemmBlocks = (N_ / 128) * (C_PAD / 128);   // 16*17 = 272

    cvt_adj<<<(N_ * N_) / 256, 256, 0, stream>>>(supports, adjB);
    pack_x0<<<CPN_ / 256, 256, 0, stream>>>(inputs, h_prev, X[0]);

    // Chebyshev pass on xs: x1 = A@x0 ; x2 = 2A@x1 - x0 ; x3 = 2A@x2 - x1
    gemm_cheb<<<gemmBlocks, 256, 0, stream>>>(adjB, X[0], nullptr, X[1], 1.f);
    gemm_cheb<<<gemmBlocks, 256, 0, stream>>>(adjB, X[1], X[0],    X[2], 2.f);
    gemm_cheb<<<gemmBlocks, 256, 0, stream>>>(adjB, X[2], X[1],    X[3], 2.f);

    gate_ru<<<BNU_ / 256, 256, 0, stream>>>(X[0], X[1], X[2], X[3],
                                            rk, rbias, uk, ubias, rBuf, uBuf);

    // Chebyshev pass on xc = concat(inputs, r*h)
    pack_xc<<<CPN_ / 256, 256, 0, stream>>>(inputs, h_prev, rBuf, X[0]);
    gemm_cheb<<<gemmBlocks, 256, 0, stream>>>(adjB, X[0], nullptr, X[1], 1.f);
    gemm_cheb<<<gemmBlocks, 256, 0, stream>>>(adjB, X[1], X[0],    X[2], 2.f);
    gemm_cheb<<<gemmBlocks, 256, 0, stream>>>(adjB, X[2], X[1],    X[3], 2.f);

    gate_c_final<<<BNU_ / 256, 256, 0, stream>>>(X[0], X[1], X[2], X[3],
                                                 ck, cbias, uBuf, h_prev, out);
}
